// HybridSelfAttentionBlock_26774826123578
// MI455X (gfx1250) — matrix operand order repaired and checked by reading, not compiled, not run on hardware
//
#include <hip/hip_runtime.h>
#include <cstdint>
#include <cstddef>

// ---------------------------------------------------------------------------
// Types for gfx1250 WMMA (wave32). bf16 operands, f32 accumulate.
// ---------------------------------------------------------------------------
typedef __bf16 bf16;
typedef bf16  v16bf __attribute__((ext_vector_type(16)));
typedef bf16  v8bf  __attribute__((ext_vector_type(8)));
typedef float v8f   __attribute__((ext_vector_type(8)));

static_assert(sizeof(bf16) == 2, "bf16 must be 2 bytes");

#define B_    2
#define LS_   2048
#define G_    32
#define KNN_  16
#define DIM_  256
#define PE_   32
#define H_    8
#define FF_   1024
#define NT_   (B_ * LS_)      // 4096 spatial tokens
#define NG_   (B_ * G_)       // 64 global tokens
#define ROWS_ (LS_ + G_)      // 2080 rows per batch in latents/output
#define NOCH  (1 << 30)       // "no remap" chunk

__device__ __forceinline__ bf16 f2bf(float f) {
  unsigned u = __builtin_bit_cast(unsigned, f);
  unsigned r = (u + 0x7FFFu + ((u >> 16) & 1u)) >> 16;  // round-to-nearest-even
  return __builtin_bit_cast(bf16, (unsigned short)r);
}

__device__ __forceinline__ float gelu_tanh(float x) {
  float u = 0.7978845608028654f * (x + 0.044715f * x * x * x);
  return 0.5f * x * (1.0f + tanhf(u));
}

__device__ __forceinline__ void load8(const float* __restrict__ p, float* r) {
  float4 a = ((const float4*)p)[0];
  float4 b = ((const float4*)p)[1];
  r[0] = a.x; r[1] = a.y; r[2] = a.z; r[3] = a.w;
  r[4] = b.x; r[5] = b.y; r[6] = b.z; r[7] = b.w;
}

// CDNA5 async global->LDS DMA (ASYNCcnt), 16B per lane per issue.
// LDS operand = low 32 bits of the generic pointer (flat LDS aperture keeps
// the 0-based LDS offset in addr[31:0]).
__device__ __forceinline__ void async_b128(const bf16* lds, const bf16* g) {
  unsigned l = (unsigned)(unsigned long long)lds;
  asm volatile("global_load_async_to_lds_b128 %0, %1, off"
               :: "v"(l), "v"(g) : "memory");
}
__device__ __forceinline__ void wait_async3() {
  asm volatile("s_wait_asynccnt 0x3" ::: "memory");
}
__device__ __forceinline__ void wait_async0() {
  asm volatile("s_wait_asynccnt 0x0" ::: "memory");
}

// ---------------------------------------------------------------------------
// Weight transpose + f32->bf16 convert: dst[n*K + k] = bf(src[k*N + n])
// ---------------------------------------------------------------------------
__global__ void k_wtrans(const float* __restrict__ src, bf16* __restrict__ dst,
                         int K, int N) {
  int i = blockIdx.x * 256 + threadIdx.x;
  if (i >= K * N) return;
  int n = i / K, k = i - n * K;
  dst[i] = f2bf(src[(size_t)k * N + n]);
}

// f32 -> bf16 elementwise with input-row remap: row = base + m + (m/chunk)*pad
__global__ void k_conv(const float* __restrict__ src, bf16* __restrict__ dst,
                       int D, int ibase, int ichunk, int ipad, int total) {
  int i = blockIdx.x * 256 + threadIdx.x;
  if (i >= total) return;
  int m = i / D, d = i - m * D;
  int row = ibase + m + (m / ichunk) * ipad;
  dst[i] = f2bf(src[(size_t)row * D + d]);
}

// ---------------------------------------------------------------------------
// LayerNorm over D=256, wave-per-row (wave32 shfl reductions), bf16 out.
// Optional second bf16 output with its own row remap (feeds the KV buffer).
// ---------------------------------------------------------------------------
__global__ void __launch_bounds__(256) k_ln(
    const float* __restrict__ in, int ibase, int ichunk, int ipad,
    const float* __restrict__ gam, const float* __restrict__ bet,
    bf16* __restrict__ out1,
    bf16* __restrict__ out2, int o2base, int o2chunk, int o2pad, int M) {
  int w = blockIdx.x * 8 + (threadIdx.x >> 5);
  int lane = threadIdx.x & 31;
  if (w >= M) return;
  int row = ibase + w + (w / ichunk) * ipad;
  float x[8];
  load8(in + (size_t)row * DIM_ + lane * 8, x);
  float s = 0.f, s2 = 0.f;
#pragma unroll
  for (int j = 0; j < 8; ++j) { s += x[j]; s2 += x[j] * x[j]; }
#pragma unroll
  for (int o = 1; o < 32; o <<= 1) {
    s  += __shfl_xor(s, o, 32);
    s2 += __shfl_xor(s2, o, 32);
  }
  float mean = s * (1.f / 256.f);
  float var  = s2 * (1.f / 256.f) - mean * mean;
  float inv  = rsqrtf(var + 1e-5f);
  int o2row = o2base + w + (w / o2chunk) * o2pad;
#pragma unroll
  for (int j = 0; j < 8; ++j) {
    int d = lane * 8 + j;
    float y = (x[j] - mean) * inv * gam[d] + bet[d];
    bf16 bv = f2bf(y);
    out1[(size_t)w * DIM_ + d] = bv;
    if (out2) out2[(size_t)o2row * DIM_ + d] = bv;
  }
}

// ---------------------------------------------------------------------------
// Generic bf16 GEMM via v_wmma_f32_16x16x32_bf16 with DOUBLE-BUFFERED
// async global->LDS staging (GLOBAL_LOAD_ASYNC_TO_LDS_B128 / ASYNCcnt).
//   C[M,N](f32) = act(A[M,K]bf16 @ Bt[N,K]^T bf16 + bias) + residual
// Block tile 128x64, 8 waves (4x2), each wave 2x2 WMMA tiles, K-step 32.
// Out-of-range rows are CLAMPED (not masked) so every wave issues exactly
// 3 async ops per tile -> deterministic s_wait_asynccnt bookkeeping; the
// garbage only reaches C rows/cols that the guarded epilogue never stores.
// Output rows remapped: crow = cbase + m + (m/cchunk)*cpad (same for Obf).
// Residual rows:         rrow = rbase + m + (m/rchunk)*rpad, ld 256.
// ---------------------------------------------------------------------------
__global__ void __launch_bounds__(256) k_gemm(
    const bf16* __restrict__ A, int lda,
    const bf16* __restrict__ Bt, int ldb,
    const float* __restrict__ bias,
    const float* __restrict__ res, int rbase, int rchunk, int rpad,
    float* __restrict__ C, int ldc, int cbase, int cchunk, int cpad,
    bf16* __restrict__ Obf, int ldo,
    int M, int N, int K, int dogelu) {
  __shared__ __align__(16) bf16 As[2][128][40];  // 40-half stride: 16B-aligned
  __shared__ __align__(16) bf16 Bs[2][64][40];

  int tid = threadIdx.x;
  int lane = tid & 31, wv = tid >> 5;
  int wm = wv >> 1, wn = wv & 1;          // 4x2 wave grid
  int m0 = blockIdx.y * 128, n0 = blockIdx.x * 64;
  int l16 = lane & 15, hi = lane >> 4;

  // Per-thread staging coordinates (row clamped to stay in-bounds).
  int ar = tid >> 1, ak = (tid & 1) << 4;          // A: 128 rows x 2x16 halfs
  int br = tid >> 2, bk = (tid & 3) << 3;          // B: 64 rows x 4x8 halfs
  int gm = m0 + ar; if (gm > M - 1) gm = M - 1;
  int gn = n0 + br; if (gn > N - 1) gn = N - 1;
  const bf16* gA = A  + (size_t)gm * lda + ak;
  const bf16* gB = Bt + (size_t)gn * ldb + bk;

  v8f acc[2][2] = {};
  union Frag { v16bf v; v8bf h[2]; };

  auto issue = [&](int buf, int k0) {
    async_b128(&As[buf][ar][ak],     gA + k0);
    async_b128(&As[buf][ar][ak + 8], gA + k0 + 8);
    async_b128(&Bs[buf][br][bk],     gB + k0);
  };

  int nsteps = K >> 5;                    // K is a multiple of 32
  issue(0, 0);
  for (int i = 0; i < nsteps; ++i) {
    int cur = i & 1;
    if (i + 1 < nsteps) {
      issue(cur ^ 1, (i + 1) << 5);
      wait_async3();                      // tile i complete (in-order returns)
    } else {
      wait_async0();
    }
    __syncthreads();                      // all waves' tile i visible in LDS

    // Fragments per ISA 7.12.2:
    //  A 16x32 bf16: lane row = l&15; halves [kb..kb+7],[kb+16..kb+23], kb=(l>>4)*8
    //  B 32x16 bf16: lane col = l&15; halves [kb..kb+7],[kb+16..kb+23], kb=(l>>4)*8
    Frag af[2], bfg[2];
    int akb = hi << 3;
    int bkb = hi << 3;
#pragma unroll
    for (int mt = 0; mt < 2; ++mt) {
      int r = (wm << 5) + (mt << 4) + l16;
      af[mt].h[0] = *(const v8bf*)&As[cur][r][akb];
      af[mt].h[1] = *(const v8bf*)&As[cur][r][akb + 16];
    }
#pragma unroll
    for (int nt = 0; nt < 2; ++nt) {
      int r = (wn << 5) + (nt << 4) + l16;
      bfg[nt].h[0] = *(const v8bf*)&Bs[cur][r][bkb];
      bfg[nt].h[1] = *(const v8bf*)&Bs[cur][r][bkb + 16];
    }
#pragma unroll
    for (int mt = 0; mt < 2; ++mt)
#pragma unroll
      for (int nt = 0; nt < 2; ++nt)
        acc[mt][nt] = __builtin_amdgcn_wmma_f32_16x16x32_bf16(
            false, af[mt].v, false, bfg[nt].v, (short)0, acc[mt][nt], false, false);
    __syncthreads();                      // reads done before buf is refilled
  }

  // Epilogue. C/D layout: lane col = l&15, rows (l>>4)*8 + r.
  int rb8 = hi << 3;
#pragma unroll
  for (int mt = 0; mt < 2; ++mt)
#pragma unroll
    for (int nt = 0; nt < 2; ++nt) {
      int cn = n0 + (wn << 5) + (nt << 4) + l16;
      if (cn >= N) continue;
      float bia = bias ? bias[cn] : 0.f;
      int gmb = m0 + (wm << 5) + (mt << 4) + rb8;
#pragma unroll
      for (int r = 0; r < 8; ++r) {
        int cm = gmb + r;
        if (cm >= M) continue;
        float v = acc[mt][nt][r] + bia;
        if (dogelu) v = gelu_tanh(v);
        if (res) v += res[(size_t)(rbase + cm + (cm / rchunk) * rpad) * DIM_ + cn];
        int crow = cbase + cm + (cm / cchunk) * cpad;
        if (C)   C[(size_t)crow * ldc + cn] = v;
        if (Obf) Obf[(size_t)crow * ldo + cn] = f2bf(v);
      }
    }
}

// ---------------------------------------------------------------------------
// Spatial attention: wave-per-token, 49 contexts (self + 16 knn + 32 global).
// K_c = Kx[idx] + Krpe[t,c] (global entries: Kg).  Bias: -d^2/(2*sigma_h^2)
// for spatial entries, sa_gbias for global entries.  f32 softmax, bf16 out.
// ---------------------------------------------------------------------------
__global__ void __launch_bounds__(256) k_sattn(
    const float* __restrict__ q,
    const float* __restrict__ kx,   const float* __restrict__ vx,
    const float* __restrict__ krpe, const float* __restrict__ vrpe,
    const float* __restrict__ kg,   const float* __restrict__ vg,
    const int* __restrict__ topk,   const float* __restrict__ dist,
    const float* __restrict__ lsig, const float* __restrict__ gbias,
    bf16* __restrict__ out) {
  __shared__ float lg[8][49][H_];
  int w = threadIdx.x >> 5, lane = threadIdx.x & 31;
  int t = blockIdx.x * 8 + w;                 // grid is exactly NT_/8 blocks
  int b = t >> 11;
  int h = lane >> 2, d0 = lane * 8;
  const float scale = 0.17677669529663687f;   // 32^-0.5

  float qv[8];
  load8(q + (size_t)t * DIM_ + d0, qv);
#pragma unroll
  for (int j = 0; j < 8; ++j) qv[j] *= scale;
  float ginv = 0.5f / expf(2.f * lsig[h]);
  float gb = gbias[0];

  for (int c = 0; c < 49; ++c) {
    const float* kp; const float* rp = nullptr; float bias;
    if (c == 0) {
      kp = kx + (size_t)t * DIM_;
      rp = krpe + (size_t)t * 17 * DIM_;
      bias = 0.f;
    } else if (c < 17) {
      int idx = topk[t * KNN_ + (c - 1)];
      kp = kx + (size_t)(b * LS_ + idx) * DIM_;
      rp = krpe + ((size_t)t * 17 + c) * DIM_;
      float dd = dist[t * KNN_ + (c - 1)];
      bias = -dd * dd * ginv;
    } else {
      kp = kg + (size_t)(b * G_ + (c - 17)) * DIM_;
      bias = gb;
    }
    float k8[8]; load8(kp + d0, k8);
    if (rp) {
      float r8[8]; load8(rp + d0, r8);
#pragma unroll
      for (int j = 0; j < 8; ++j) k8[j] += r8[j];
    }
    float dot = 0.f;
#pragma unroll
    for (int j = 0; j < 8; ++j) dot += qv[j] * k8[j];
    dot += __shfl_xor(dot, 1, 32);
    dot += __shfl_xor(dot, 2, 32);            // reduce 4-lane head group
    if ((lane & 3) == 0) lg[w][c][h] = dot + bias;
  }
  __syncthreads();

  float mx = -1e30f;
  for (int c = 0; c < 49; ++c) mx = fmaxf(mx, lg[w][c][h]);
  float ssum = 0.f;
  for (int c = 0; c < 49; ++c) ssum += expf(lg[w][c][h] - mx);
  float inv = 1.f / ssum;

  float acc[8] = {0, 0, 0, 0, 0, 0, 0, 0};
  for (int c = 0; c < 49; ++c) {
    float wt = expf(lg[w][c][h] - mx) * inv;
    const float* vp; const float* rp = nullptr;
    if (c == 0) {
      vp = vx + (size_t)t * DIM_;
      rp = vrpe + (size_t)t * 17 * DIM_;
    } else if (c < 17) {
      int idx = topk[t * KNN_ + (c - 1)];
      vp = vx + (size_t)(b * LS_ + idx) * DIM_;
      rp = vrpe + ((size_t)t * 17 + c) * DIM_;
    } else {
      vp = vg + (size_t)(b * G_ + (c - 17)) * DIM_;
    }
    float v8a[8]; load8(vp + d0, v8a);
    if (rp) {
      float r8[8]; load8(rp + d0, r8);
#pragma unroll
      for (int j = 0; j < 8; ++j) v8a[j] += r8[j];
    }
#pragma unroll
    for (int j = 0; j < 8; ++j) acc[j] += wt * v8a[j];
  }
#pragma unroll
  for (int j = 0; j < 8; ++j) out[(size_t)t * DIM_ + d0 + j] = f2bf(acc[j]);
}

// ---------------------------------------------------------------------------
// Global attention: wave-per-(b,g) query, online softmax over 2080 keys.
// (ga_gbias is a scalar added to every logit -> softmax-invariant, dropped.)
// ---------------------------------------------------------------------------
__global__ void __launch_bounds__(256) k_gattn(
    const float* __restrict__ qg, const float* __restrict__ kk,
    const float* __restrict__ vv, bf16* __restrict__ og) {
  int w = (blockIdx.x * 256 + threadIdx.x) >> 5;
  int lane = threadIdx.x & 31;
  if (w >= NG_) return;
  int b = w >> 5;
  int d0 = lane * 8;
  const float scale = 0.17677669529663687f;

  float qv[8];
  load8(qg + (size_t)w * DIM_ + d0, qv);
#pragma unroll
  for (int j = 0; j < 8; ++j) qv[j] *= scale;

  float m = -1e30f, s = 0.f;
  float acc[8] = {0, 0, 0, 0, 0, 0, 0, 0};
  size_t base = (size_t)b * ROWS_ * DIM_;
  for (int c = 0; c < ROWS_; ++c) {
    float k8[8]; load8(kk + base + (size_t)c * DIM_ + d0, k8);
    float dot = 0.f;
#pragma unroll
    for (int j = 0; j < 8; ++j) dot += qv[j] * k8[j];
    dot += __shfl_xor(dot, 1, 32);
    dot += __shfl_xor(dot, 2, 32);
    float nm = fmaxf(m, dot);
    float f = expf(m - nm);
    float e = expf(dot - nm);
    s = s * f + e;
    float v8a[8]; load8(vv + base + (size_t)c * DIM_ + d0, v8a);
#pragma unroll
    for (int j = 0; j < 8; ++j) acc[j] = acc[j] * f + e * v8a[j];
    m = nm;
  }
  float inv = 1.f / s;
#pragma unroll
  for (int j = 0; j < 8; ++j) og[(size_t)w * DIM_ + d0 + j] = f2bf(acc[j] * inv);
}

// ---------------------------------------------------------------------------
// Host driver
// ---------------------------------------------------------------------------
extern "C" void kernel_launch(void* const* d_in, const int* in_sizes, int n_in,
                              void* d_out, int out_size, void* d_ws, size_t ws_size,
                              hipStream_t stream) {
  (void)in_sizes; (void)n_in; (void)out_size; (void)ws_size;
  const float* latents = (const float*)d_in[0];
  const int*   topk    = (const int*)d_in[1];
  const float* rpe     = (const float*)d_in[2];
  const float* srpe    = (const float*)d_in[3];
  const float* dist    = (const float*)d_in[4];
  // d_in[5] = num_spatial (compile-time constant LS_)
  const float* sa_wq   = (const float*)d_in[6];
  const float* sa_wk   = (const float*)d_in[7];
  const float* sa_wv   = (const float*)d_in[8];
  const float* sa_wo   = (const float*)d_in[9];
  const float* sa_bo   = (const float*)d_in[10];
  const float* sa_lsig = (const float*)d_in[11];
  const float* sa_gb   = (const float*)d_in[12];
  const float* ln1_g   = (const float*)d_in[13];
  const float* ln1_b   = (const float*)d_in[14];
  const float* ln2_g   = (const float*)d_in[15];
  const float* ln2_b   = (const float*)d_in[16];
  const float* sff_w1  = (const float*)d_in[17];
  const float* sff_b1  = (const float*)d_in[18];
  const float* sff_w2  = (const float*)d_in[19];
  const float* sff_b2  = (const float*)d_in[20];
  const float* ga_wq   = (const float*)d_in[21];
  const float* ga_wk   = (const float*)d_in[22];
  const float* ga_wv   = (const float*)d_in[23];
  const float* ga_wo   = (const float*)d_in[24];
  const float* ga_bo   = (const float*)d_in[25];
  // d_in[26] = ga_gbias (softmax-invariant scalar, unused)
  const float* ln3_g   = (const float*)d_in[27];
  const float* ln3_b   = (const float*)d_in[28];
  const float* ln4_g   = (const float*)d_in[29];
  const float* ln4_b   = (const float*)d_in[30];
  const float* gff_w1  = (const float*)d_in[31];
  const float* gff_b1  = (const float*)d_in[32];
  const float* gff_w2  = (const float*)d_in[33];
  const float* gff_b2  = (const float*)d_in[34];
  float* out = (float*)d_out;

  char* wsb = (char*)d_ws;
  size_t off = 0;
  auto alloc = [&](size_t bytes) -> void* {
    void* p = wsb + off;
    off = (off + bytes + 255) & ~(size_t)255;
    return p;
  };
  // transposed bf16 weights (N rows x K cols)
  bf16* wq_t  = (bf16*)alloc((size_t)256 * 256 * 2);
  bf16* wk_t  = (bf16*)alloc((size_t)256 * 288 * 2);   // k<256: wk_x, k>=256: wk_pe
  bf16* wv_t  = (bf16*)alloc((size_t)256 * 288 * 2);
  bf16* wo_t  = (bf16*)alloc((size_t)256 * 256 * 2);
  bf16* w1s_t = (bf16*)alloc((size_t)1024 * 256 * 2);
  bf16* w2s_t = (bf16*)alloc((size_t)256 * 1024 * 2);
  bf16* gwq_t = (bf16*)alloc((size_t)256 * 256 * 2);
  bf16* gwk_t = (bf16*)alloc((size_t)256 * 256 * 2);
  bf16* gwv_t = (bf16*)alloc((size_t)256 * 256 * 2);
  bf16* gwo_t = (bf16*)alloc((size_t)256 * 256 * 2);
  bf16* w1g_t = (bf16*)alloc((size_t)1024 * 256 * 2);
  bf16* w2g_t = (bf16*)alloc((size_t)256 * 1024 * 2);
  // activations
  bf16*  x_bf    = (bf16*)alloc((size_t)NT_ * 256 * 2);
  bf16*  glob_bf = (bf16*)alloc((size_t)NG_ * 256 * 2);
  bf16*  rpe_bf  = (bf16*)alloc((size_t)NT_ * KNN_ * 32 * 2);
  bf16*  srpe_bf = (bf16*)alloc((size_t)NT_ * 32 * 2);
  float* q_f     = (float*)alloc((size_t)NT_ * 256 * 4);
  float* kx_f    = (float*)alloc((size_t)NT_ * 256 * 4);
  float* vx_f    = (float*)alloc((size_t)NT_ * 256 * 4);
  float* kg_f    = (float*)alloc((size_t)NG_ * 256 * 4);
  float* vg_f    = (float*)alloc((size_t)NG_ * 256 * 4);
  float* krpe_f  = (float*)alloc((size_t)NT_ * 17 * 256 * 4);
  float* vrpe_f  = (float*)alloc((size_t)NT_ * 17 * 256 * 4);
  bf16*  ao_bf   = (bf16*)alloc((size_t)NT_ * 256 * 2);
  float* sp1_f   = (float*)alloc((size_t)NT_ * 256 * 4);
  bf16*  x2_bf   = (bf16*)alloc((size_t)NT_ * 256 * 2);
  bf16*  h1_bf   = (bf16*)alloc((size_t)NT_ * 1024 * 2);
  bf16*  kv_bf   = (bf16*)alloc((size_t)B_ * ROWS_ * 256 * 2);
  bf16*  gq_bf   = (bf16*)alloc((size_t)NG_ * 256 * 2);
  float* qg_f    = (float*)alloc((size_t)NG_ * 256 * 4);
  float* kg2_f   = (float*)alloc((size_t)B_ * ROWS_ * 256 * 4);
  float* vg2_f   = (float*)alloc((size_t)B_ * ROWS_ * 256 * 4);
  bf16*  og_bf   = (bf16*)alloc((size_t)NG_ * 256 * 2);
  float* gl1_f   = (float*)alloc((size_t)NG_ * 256 * 4);
  bf16*  x4_bf   = (bf16*)alloc((size_t)NG_ * 256 * 2);
  bf16*  h2_bf   = (bf16*)alloc((size_t)NG_ * 1024 * 2);

  auto WT = [&](const float* src, bf16* dst, int K, int N) {
    int tot = K * N;
    k_wtrans<<<(tot + 255) / 256, 256, 0, stream>>>(src, dst, K, N);
  };
  auto GEMM = [&](const bf16* A, int lda, const bf16* Bt, int ldb,
                  const float* bias, const float* res, int rb, int rc, int rp,
                  float* C, int ldc, int cb, int cc, int cp,
                  bf16* Obf, int ldo, int M, int N, int K, int gel) {
    dim3 g((N + 63) / 64, (M + 127) / 128);
    k_gemm<<<g, 256, 0, stream>>>(A, lda, Bt, ldb, bias, res, rb, rc, rp,
                                  C, ldc, cb, cc, cp, Obf, ldo, M, N, K, gel);
  };

  // --- weight prep ---------------------------------------------------------
  WT(sa_wq, wq_t, 256, 256);   WT(sa_wk, wk_t, 288, 256);
  WT(sa_wv, wv_t, 288, 256);   WT(sa_wo, wo_t, 256, 256);
  WT(sff_w1, w1s_t, 256, 1024); WT(sff_w2, w2s_t, 1024, 256);
  WT(ga_wq, gwq_t, 256, 256);  WT(ga_wk, gwk_t, 256, 256);
  WT(ga_wv, gwv_t, 256, 256);  WT(ga_wo, gwo_t, 256, 256);
  WT(gff_w1, w1g_t, 256, 1024); WT(gff_w2, w2g_t, 1024, 256);

  // --- bf16 converts -------------------------------------------------------
  k_conv<<<(NG_ * 256 + 255) / 256, 256, 0, stream>>>(latents, glob_bf, 256, 2048, 32, 2048, NG_ * 256);
  k_conv<<<(NT_ * KNN_ * 32 + 255) / 256, 256, 0, stream>>>(rpe, rpe_bf, 32, 0, NOCH, 0, NT_ * KNN_ * 32);
  k_conv<<<(NT_ * 32 + 255) / 256, 256, 0, stream>>>(srpe, srpe_bf, 32, 0, NOCH, 0, NT_ * 32);

  // --- spatial branch ------------------------------------------------------
  k_ln<<<NT_ / 8, 256, 0, stream>>>(latents, 0, 2048, 32, ln1_g, ln1_b,
                                    x_bf, nullptr, 0, NOCH, 0, NT_);
  GEMM(x_bf, 256, wq_t, 256, nullptr, nullptr, 0, 1, 0, q_f, 256, 0, NOCH, 0, nullptr, 0, NT_, 256, 256, 0);
  GEMM(x_bf, 256, wk_t, 288, nullptr, nullptr, 0, 1, 0, kx_f, 256, 0, NOCH, 0, nullptr, 0, NT_, 256, 256, 0);
  GEMM(x_bf, 256, wv_t, 288, nullptr, nullptr, 0, 1, 0, vx_f, 256, 0, NOCH, 0, nullptr, 0, NT_, 256, 256, 0);
  GEMM(glob_bf, 256, wk_t, 288, nullptr, nullptr, 0, 1, 0, kg_f, 256, 0, NOCH, 0, nullptr, 0, NG_, 256, 256, 0);
  GEMM(glob_bf, 256, wv_t, 288, nullptr, nullptr, 0, 1, 0, vg_f, 256, 0, NOCH, 0, nullptr, 0, NG_, 256, 256, 0);
  // rpe projections through last 32 weight rows; scatter into (t,17,256)
  GEMM(rpe_bf, 32, wk_t + 256, 288, nullptr, nullptr, 0, 1, 0, krpe_f, 256, 1, 16, 1, nullptr, 0, NT_ * KNN_, 256, 32, 0);
  GEMM(rpe_bf, 32, wv_t + 256, 288, nullptr, nullptr, 0, 1, 0, vrpe_f, 256, 1, 16, 1, nullptr, 0, NT_ * KNN_, 256, 32, 0);
  GEMM(srpe_bf, 32, wk_t + 256, 288, nullptr, nullptr, 0, 1, 0, krpe_f, 256, 0, 1, 16, nullptr, 0, NT_, 256, 32, 0);
  GEMM(srpe_bf, 32, wv_t + 256, 288, nullptr, nullptr, 0, 1, 0, vrpe_f, 256, 0, 1, 16, nullptr, 0, NT_, 256, 32, 0);

  k_sattn<<<NT_ / 8, 256, 0, stream>>>(q_f, kx_f, vx_f, krpe_f, vrpe_f, kg_f, vg_f,
                                       topk, dist, sa_lsig, sa_gb, ao_bf);

  // out-proj + residual(latents spatial rows)
  GEMM(ao_bf, 256, wo_t, 256, sa_bo, latents, 0, 2048, 32, sp1_f, 256, 0, NOCH, 0, nullptr, 0, NT_, 256, 256, 0);
  // spatial FF
  k_ln<<<NT_ / 8, 256, 0, stream>>>(sp1_f, 0, NOCH, 0, ln2_g, ln2_b,
                                    x2_bf, nullptr, 0, NOCH, 0, NT_);
  GEMM(x2_bf, 256, w1s_t, 256, sff_b1, nullptr, 0, 1, 0, nullptr, 0, 0, NOCH, 0, h1_bf, 1024, NT_, 1024, 256, 1);
  // second FF GEMM writes final spatial output rows AND bf16 copy into KV buf
  GEMM(h1_bf, 1024, w2s_t, 1024, sff_b2, sp1_f, 0, NOCH, 0, out, 256, 0, 2048, 32, kv_bf, 256, NT_, 256, 1024, 0);

  // --- global branch -------------------------------------------------------
  k_ln<<<(NG_ + 7) / 8, 256, 0, stream>>>(latents, 2048, 32, 2048, ln3_g, ln3_b,
                                          gq_bf, kv_bf, 2048, 32, 2048, NG_);
  GEMM(gq_bf, 256, gwq_t, 256, nullptr, nullptr, 0, 1, 0, qg_f, 256, 0, NOCH, 0, nullptr, 0, NG_, 256, 256, 0);
  GEMM(kv_bf, 256, gwk_t, 256, nullptr, nullptr, 0, 1, 0, kg2_f, 256, 0, NOCH, 0, nullptr, 0, B_ * ROWS_, 256, 256, 0);
  GEMM(kv_bf, 256, gwv_t, 256, nullptr, nullptr, 0, 1, 0, vg2_f, 256, 0, NOCH, 0, nullptr, 0, B_ * ROWS_, 256, 256, 0);

  k_gattn<<<(NG_ * 32 + 255) / 256, 256, 0, stream>>>(qg_f, kg2_f, vg2_f, og_bf);

  GEMM(og_bf, 256, gwo_t, 256, ga_bo, latents, 2048, 32, 2048, gl1_f, 256, 0, NOCH, 0, nullptr, 0, NG_, 256, 256, 0);
  k_ln<<<(NG_ + 7) / 8, 256, 0, stream>>>(gl1_f, 0, NOCH, 0, ln4_g, ln4_b,
                                          x4_bf, nullptr, 0, NOCH, 0, NG_);
  GEMM(x4_bf, 256, w1g_t, 256, gff_b1, nullptr, 0, 1, 0, nullptr, 0, 0, NOCH, 0, h2_bf, 1024, NG_, 1024, 256, 1);
  // final global output rows of d_out
  GEMM(h2_bf, 1024, w2g_t, 1024, gff_b2, gl1_f, 0, NOCH, 0, out, 256, 2048, 32, 2048, nullptr, 0, NG_, 256, 1024, 0);
}
